// SpatialDistanceGraph_58420145160345
// MI455X (gfx1250) — matrix operand order repaired and checked by reading, not compiled, not run on hardware
//
#include <hip/hip_runtime.h>
#include <hip/hip_bf16.h>
#include <math.h>

// ---------------------------------------------------------------------------
// CDNA5 (gfx1250) implementation of SpatialDistanceGraph forward.
//
// Sinkhorn restructuring: since the reference adds K (=exp(-C/eps)) inside
// logsumexp, exp factors:  logsumexp_j(v_j + K_ij) = m + log( (F t)_i ),
// F = exp(K) in (1,e] (f16-safe), t_j = exp(v_j - m) in [0,1].
// Both half-updates reduce over K's 2nd index -> both are F * vector.
// Batch 16 problems per workgroup -> F[240x256(pad)] x T[256x16] per step,
// mapped to v_wmma_f32_16x16x32_f16 with f32 accumulation.
// ---------------------------------------------------------------------------

typedef __attribute__((ext_vector_type(16))) _Float16 v16h;
typedef __attribute__((ext_vector_type(8)))  float    v8f;

#define NPROB   728            // 8 batches * 91 pairs
#define PROBPAD 736            // 46 workgroups * 16
#define SINK_ITERS 100

__device__ __forceinline__ void unrank_pair(int pr, int& i0, int& j0) {
  int i = 0, rem = pr;
  while (rem >= 13 - i) { rem -= 13 - i; ++i; }
  i0 = i; j0 = i + 1 + rem;
}

// A-fragment (16-bit A 16x32) k-offset within a 32-wide K step, contiguous order (k = 16*khalf + e).
__device__ __forceinline__ int a_koff(int e, int khalf) {
  return (khalf << 4) + e;
}

// ---------------------------------------------------------------------------
// Prep: cost matrix -> F = exp(exp(-10*C)) packed in WMMA A-fragment order,
// plus G = exp(K)*C for the EMD epilogue.
// ---------------------------------------------------------------------------
__device__ __forceinline__ float cost_at(int pI, int qI, int gs, float cmax,
                                         int tthird, float a0, float a1, float a2) {
  int py = pI / gs, px = pI - py * gs;
  int qy = qI / gs, qx = qI - qy * gs;
  float dx = (float)(px - qx), dy = (float)(py - qy);
  float c = sqrtf(dx * dx + dy * dy) / cmax;
  float w = (py < tthird) ? a0 : (py < 2 * tthird) ? a1 : a2;
  return c * w;
}

__global__ void prep_cost_kernel(_Float16* __restrict__ Efrag,
                                 float* __restrict__ G,
                                 const float* __restrict__ anat_w,
                                 int gs, int n, int Mtiles, int Ksteps) {
  int fragElems = Mtiles * Ksteps * 32 * 16;
  int total = fragElems + n * n;
  int idx = blockIdx.x * 256 + threadIdx.x;
  if (idx >= total) return;
  float a0 = anat_w[0], a1 = anat_w[1], a2 = anat_w[2];
  float cmax = sqrtf(2.f) * (float)(gs - 1);
  int tthird = gs / 3;
  if (idx < fragElems) {
    int e = idx & 15;
    int lane2 = (idx >> 4) & 31;
    int ksmt = idx >> 9;                 // mt*Ksteps + ks
    int ks = ksmt % Ksteps, mt = ksmt / Ksteps;
    int m = mt * 16 + (lane2 & 15);
    int khalf = lane2 >> 4;
    int k = ks * 32 + a_koff(e, khalf);
    float val = 0.f;
    if (m < n && k < n) {
      float C = cost_at(m, k, gs, cmax, tthird, a0, a1, a2);
      val = __expf(__expf(-10.f * C));   // F = exp(K), K = exp(-C/0.1)
    }
    Efrag[idx] = (_Float16)val;
  } else {
    int r = idx - fragElems;
    int i = r / n, j = r - i * n;
    float C = cost_at(i, j, gs, cmax, tthird, a0, a1, a2);
    G[r] = __expf(__expf(-10.f * C)) * C;  // e^{K_ij} * C_ij
  }
}

// ---------------------------------------------------------------------------
// Prep: spatial distributions -> logP/logQ laid out [PROBPAD][rows]
// ---------------------------------------------------------------------------
__device__ float dist_val(const float* bb, int b, int d, int gs, int xx, int yy) {
  const float* box = bb + (size_t)(b * 14 + d) * 4;
  float x = box[0], y = box[1], w = box[2], h = box[3];
  bool anyNz = (x != 0.f) || (y != 0.f) || (w != 0.f) || (h != 0.f);
  float sc = (float)gs / 1000.f;
  float cx = x * sc, cy = y * sc, cw = w * sc, ch = h * sc;
  float x1 = fmaxf(0.f, truncf(cx)), y1 = fmaxf(0.f, truncf(cy));
  float x2 = fminf((float)gs - 1.f, truncf(cx + cw));
  float y2 = fminf((float)gs - 1.f, truncf(cy + ch));
  bool valid = anyNz && (x1 < x2) && (y1 < y2);
  bool in = valid && ((float)yy >= y1) && ((float)yy <= y2) &&
                     ((float)xx >= x1) && ((float)xx <= x2);
  return in ? 1.f : 0.f;
}

__global__ void prep_logpq_kernel(float* __restrict__ logP, float* __restrict__ logQ,
                                  const float* __restrict__ bb, int gs, int n, int rows) {
  int idx = blockIdx.x * 256 + threadIdx.x;
  int total = PROBPAD * rows;
  if (idx >= total) return;
  int prob = idx / rows, m = idx - prob * rows;
  float lp = 0.f, lq = 0.f;
  if (prob < NPROB && m < n) {
    int b = prob / 91, pr = prob - b * 91;
    int i0, j0; unrank_pair(pr, i0, j0);
    int yy = m / gs, xx = m - yy * gs;
    lp = logf(dist_val(bb, b, i0, gs, xx, yy) + 1e-8f);
    lq = logf(dist_val(bb, b, j0, gs, xx, yy) + 1e-8f);
  }
  logP[idx] = lp;
  logQ[idx] = lq;
}

// ---------------------------------------------------------------------------
// Sinkhorn: one workgroup (256 thr = 8 wave32) handles 16 problems for 100
// iterations, locally in LDS. GEMM via v_wmma_f32_16x16x32_f16.
// ---------------------------------------------------------------------------
__device__ __forceinline__ void col_max(float (*src)[240], float* mx,
                                        float (*red)[16], int n, int tid) {
  int p = tid >> 4, s = tid & 15;
  float m = -1e30f;
  for (int kk = s; kk < n; kk += 16) m = fmaxf(m, src[p][kk]);
  red[p][s] = m;
  __syncthreads();
  if (tid < 16) {
    float mm = red[tid][0];
#pragma unroll
    for (int q = 1; q < 16; ++q) mm = fmaxf(mm, red[tid][q]);
    mx[tid] = mm;
  }
  __syncthreads();
}

__device__ __forceinline__ void half_update(
    float (*src)[240], float (*dst)[240], float* mx,
    _Float16 (*ts)[256], float (*red)[16],
    const _Float16* __restrict__ Efrag, const float* __restrict__ logPQ,
    int n, int Mtiles, int Ksteps, int rows, int Kpad, int probBase,
    int tid, int lane, int wave) {
  col_max(src, mx, red, n, tid);
  for (int idx = tid; idx < 16 * Kpad; idx += 256) {
    int pp = idx / Kpad, kk = idx - pp * Kpad;
    float val = (kk < n) ? __expf(src[pp][kk] - mx[pp]) : 0.f;
    ts[pp][kk] = (_Float16)val;
  }
  __syncthreads();
  const v16h* Ef = (const v16h*)Efrag;
  int ncol = lane & 15;
  int khalf = lane >> 4;
  for (int mt = wave; mt < Mtiles; mt += 8) {
    v8f acc = {};
    for (int ks = 0; ks < Ksteps; ++ks) {
      v16h a = Ef[(mt * Ksteps + ks) * 32 + lane];
      v16h bf = *(const v16h*)&ts[ncol][ks * 32 + khalf * 16];
      acc = __builtin_amdgcn_wmma_f32_16x16x32_f16(false, a, false, bf,
                                                   (short)0, acc, false, false);
    }
    int mbase = mt * 16 + khalf * 8;
    const float* lp = logPQ + (size_t)(probBase + ncol) * rows;
#pragma unroll
    for (int r = 0; r < 8; ++r) {
      int mrow = mbase + r;
      if (mrow < n) dst[ncol][mrow] = lp[mrow] - mx[ncol] - __logf(acc[r]);
    }
  }
  __syncthreads();
}

__global__ void sinkhorn_kernel(const _Float16* __restrict__ Efrag,
                                const float* __restrict__ logP,
                                const float* __restrict__ logQ,
                                const float* __restrict__ G,
                                float* __restrict__ adj,
                                int n, int Mtiles, int Ksteps, float sw) {
  __shared__ float u_s[16][240];
  __shared__ float v_s[16][240];
  __shared__ float w_s[16][240];
  __shared__ __align__(32) _Float16 t_s[16][256];
  __shared__ float red_s[16][16];
  __shared__ float mxv[16];
  __shared__ float mxu[16];

  int tid = threadIdx.x, lane = tid & 31, wave = tid >> 5;
  int probBase = blockIdx.x * 16;
  int rows = Mtiles * 16, Kpad = Ksteps * 32;

  for (int idx = tid; idx < 16 * rows; idx += 256) {
    int pp = idx / rows, rr = idx - pp * rows;
    u_s[pp][rr] = 0.f;
    v_s[pp][rr] = 0.f;
  }
  __syncthreads();

  for (int it = 0; it < SINK_ITERS; ++it) {
    half_update(v_s, u_s, mxv, t_s, red_s, Efrag, logP,
                n, Mtiles, Ksteps, rows, Kpad, probBase, tid, lane, wave);
    half_update(u_s, v_s, mxu, t_s, red_s, Efrag, logQ,
                n, Mtiles, Ksteps, rows, Kpad, probBase, tid, lane, wave);
  }

  // EMD epilogue: emd = e^{mu+mv} * sum_i e^{u_i-mu} * sum_j G_ij e^{v_j-mv}
  col_max(u_s, mxu, red_s, n, tid);
  col_max(v_s, mxv, red_s, n, tid);
  for (int idx = tid; idx < 16 * rows; idx += 256) {
    int pp = idx / rows, jj = idx - pp * rows;
    w_s[pp][jj] = (jj < n) ? __expf(v_s[pp][jj] - mxv[pp]) : 0.f;
  }
  __syncthreads();

  for (int p = wave; p < 16; p += 8) {
    int prob = probBase + p;
    if (prob >= NPROB) continue;
    float acc = 0.f;
    for (int i = lane; i < n; i += 32) {
      const float* Gi = G + (size_t)i * n;
      float si = 0.f;
      for (int j = 0; j < n; ++j) si += Gi[j] * w_s[p][j];
      acc += __expf(u_s[p][i] - mxu[p]) * si;
    }
#pragma unroll
    for (int off = 16; off > 0; off >>= 1) acc += __shfl_down(acc, off);
    if (lane == 0) {
      float emd = acc * __expf(mxu[p] + mxv[p]);
      int b = prob / 91, pr = prob - b * 91;
      int i0, j0; unrank_pair(pr, i0, j0);
      float valw = sw * emd;
      adj[(b * 14 + i0) * 14 + j0] += valw;   // unique per thread per launch
      adj[(b * 14 + j0) * 14 + i0] += valw;
    }
  }
}

// ---------------------------------------------------------------------------
// Generic WMMA GEMM: C[M,N] = A[M,K] @ W[K,N] + bias.  One wave per 16x16 tile.
// M % 16 == 0, K % 32 == 0, N % 16 == 0 (true for 112 / {256,768}).
// ---------------------------------------------------------------------------
__global__ void gemm_wmma_kernel(const float* __restrict__ A, int lda,
                                 const float* __restrict__ W, int ldw,
                                 const float* __restrict__ bias,
                                 float* __restrict__ C, int ldc, int K) {
  int mt = blockIdx.x, nt = blockIdx.y;
  int lane = threadIdx.x & 31;
  int mrow = mt * 16 + (lane & 15);
  int khalf = lane >> 4;
  int ncol = nt * 16 + (lane & 15);
  v8f acc = {};
  for (int kb = 0; kb < K; kb += 32) {
    v16h a, bf;
#pragma unroll
    for (int e = 0; e < 16; ++e)
      a[e] = (_Float16)A[(size_t)mrow * lda + kb + a_koff(e, khalf)];
    int kb2 = kb + khalf * 16;
#pragma unroll
    for (int e = 0; e < 16; ++e)
      bf[e] = (_Float16)W[(size_t)(kb2 + e) * ldw + ncol];
    acc = __builtin_amdgcn_wmma_f32_16x16x32_f16(false, a, false, bf,
                                                 (short)0, acc, false, false);
  }
  int mbase = mt * 16 + khalf * 8;
  float bv = bias[ncol];
#pragma unroll
  for (int r = 0; r < 8; ++r)
    C[(size_t)(mbase + r) * ldc + ncol] = acc[r] + bv;
}

// ---------------------------------------------------------------------------
// LayerNorm (optionally fused ELU). One 256-thread block per row.
// ---------------------------------------------------------------------------
__global__ void ln_kernel(const float* __restrict__ X,
                          const float* __restrict__ g, const float* __restrict__ b,
                          float* __restrict__ Y, int dim, int doElu) {
  __shared__ float r1[256];
  __shared__ float r2[256];
  int row = blockIdx.x, t = threadIdx.x;
  float s = 0.f, s2 = 0.f;
  for (int c = t; c < dim; c += 256) {
    float x = X[(size_t)row * dim + c];
    s += x; s2 += x * x;
  }
  r1[t] = s; r2[t] = s2;
  __syncthreads();
  for (int off = 128; off > 0; off >>= 1) {
    if (t < off) { r1[t] += r1[t + off]; r2[t] += r2[t + off]; }
    __syncthreads();
  }
  float mean = r1[0] / (float)dim;
  float var = r2[0] / (float)dim - mean * mean;
  float inv = rsqrtf(var + 1e-5f);
  for (int c = t; c < dim; c += 256) {
    float y = (X[(size_t)row * dim + c] - mean) * inv * g[c] + b[c];
    if (doElu) y = (y > 0.f) ? y : expm1f(y);
    Y[(size_t)row * dim + c] = y;
  }
}

// ---------------------------------------------------------------------------
// adj softmax (rows of 14)
// ---------------------------------------------------------------------------
__global__ void adj_softmax_kernel(const float* __restrict__ adj,
                                   float* __restrict__ adjS) {
  int t = blockIdx.x * 128 + threadIdx.x;
  if (t >= 112) return;
  const float* a = adj + (size_t)t * 14;
  float* o = adjS + (size_t)t * 14;
  float mx = -1e30f;
  for (int j = 0; j < 14; ++j) mx = fmaxf(mx, a[j]);
  float sum = 0.f;
  float e[14];
  for (int j = 0; j < 14; ++j) { e[j] = __expf(a[j] - mx); sum += e[j]; }
  float inv = 1.f / sum;
  for (int j = 0; j < 14; ++j) o[j] = e[j] * inv;
}

// ---------------------------------------------------------------------------
// GAT attention (8 heads, 14 nodes): one block per batch element.
// Also accumulates attn/24 into attnsum (mean over 8 heads and 3 layers).
// ---------------------------------------------------------------------------
__global__ void gat_attention_kernel(const float* __restrict__ q,
                                     const float* __restrict__ k,
                                     const float* __restrict__ v,
                                     const float* __restrict__ adjS,
                                     float* __restrict__ attno,
                                     float* __restrict__ attnsum) {
  __shared__ float att[8][14][14];
  int b = blockIdx.x, t = threadIdx.x;
  const float iscale = 0.17677669529663687f;  // 1/sqrt(32)
  if (t < 112) {
    int h = t / 14, i = t % 14;
    float s[14];
    float mx = -1e30f;
    for (int j = 0; j < 14; ++j) {
      float dot = 0.f;
      const float* qi = q + (size_t)(b * 14 + i) * 256 + h * 32;
      const float* kj = k + (size_t)(b * 14 + j) * 256 + h * 32;
#pragma unroll
      for (int d = 0; d < 32; ++d) dot += qi[d] * kj[d];
      dot *= iscale;
      if (adjS[(b * 14 + i) * 14 + j] == 0.f) dot = -1e30f;
      s[j] = dot;
      mx = fmaxf(mx, dot);
    }
    float sum = 0.f;
    for (int j = 0; j < 14; ++j) { s[j] = __expf(s[j] - mx); sum += s[j]; }
    float inv = 1.f / sum;
    for (int j = 0; j < 14; ++j) { s[j] *= inv; att[h][i][j] = s[j]; }
    float* oi = attno + (size_t)(b * 14 + i) * 256 + h * 32;
#pragma unroll
    for (int d = 0; d < 32; ++d) {
      float o = 0.f;
      for (int j = 0; j < 14; ++j) o += s[j] * v[(size_t)(b * 14 + j) * 256 + h * 32 + d];
      oi[d] = o;
    }
  }
  __syncthreads();
  if (t < 196) {
    int i = t / 14, j = t % 14;
    float acc = 0.f;
#pragma unroll
    for (int h = 0; h < 8; ++h) acc += att[h][i][j];
    attnsum[(b * 14 + i) * 14 + j] += acc * (1.f / 24.f);
  }
}

// ---------------------------------------------------------------------------
// small utility kernels
// ---------------------------------------------------------------------------
__global__ void zero_kernel(float* a, int n) {
  int t = blockIdx.x * 256 + threadIdx.x;
  if (t < n) a[t] = 0.f;
}
__global__ void copy_kernel(const float* __restrict__ src, float* __restrict__ dst, int n) {
  int t = blockIdx.x * 256 + threadIdx.x;
  if (t < n) dst[t] = src[t];
}

// ---------------------------------------------------------------------------
// Host launcher
// ---------------------------------------------------------------------------
extern "C" void kernel_launch(void* const* d_in, const int* in_sizes, int n_in,
                              void* d_out, int out_size, void* d_ws, size_t ws_size,
                              hipStream_t stream) {
  (void)in_sizes; (void)n_in; (void)out_size; (void)ws_size;
  // Input order (setup_inputs() insertion order, nested dicts DFS):
  const float* features = (const float*)d_in[0];   // [8,14,768]
  const float* bb       = (const float*)d_in[1];   // [8,14,4]
  const float* embw = (const float*)d_in[2];
  const float* embb = (const float*)d_in[3];
  const float* n1g  = (const float*)d_in[4];
  const float* n1b  = (const float*)d_in[5];
  // gats[L]: base 6 + L*10 : wq.w,wq.b,wk.w,wk.b,wv.w,wv.b,wo.w,wo.b,ln.g,ln.b
  const float* projw = (const float*)d_in[36];
  const float* projb = (const float*)d_in[37];
  const float* n2g   = (const float*)d_in[38];
  const float* n2b   = (const float*)d_in[39];
  const float* anat  = (const float*)d_in[40];

  // carve workspace (256B aligned regions)
  char* base = (char*)d_ws;
  size_t off = 0;
  auto carve = [&](size_t bytes) -> char* {
    size_t start = (off + 255) & ~(size_t)255;
    off = start + bytes;
    return base + start;
  };
  float*    adj     = (float*)carve(1568 * 4);
  float*    adjS    = (float*)carve(1568 * 4);
  float*    attnsum = (float*)carve(1568 * 4);
  _Float16* Ef15    = (_Float16*)carve(15 * 8 * 32 * 16 * 2);
  _Float16* Ef5     = (_Float16*)carve(2 * 1 * 32 * 16 * 2);
  float*    G15     = (float*)carve(225 * 225 * 4);
  float*    G5      = (float*)carve(25 * 25 * 4);
  float*    logP15  = (float*)carve((size_t)PROBPAD * 240 * 4);
  float*    logQ15  = (float*)carve((size_t)PROBPAD * 240 * 4);
  float*    logP5   = (float*)carve((size_t)PROBPAD * 32 * 4);
  float*    logQ5   = (float*)carve((size_t)PROBPAD * 32 * 4);
  float*    xbuf    = (float*)carve(112 * 256 * 4);
  float*    t256    = (float*)carve(112 * 256 * 4);
  float*    qb      = (float*)carve(112 * 256 * 4);
  float*    kb      = (float*)carve(112 * 256 * 4);
  float*    vb      = (float*)carve(112 * 256 * 4);
  float*    attno   = (float*)carve(112 * 256 * 4);
  float*    t768    = (float*)carve(112 * 768 * 4);

  float* out_main = (float*)d_out;             // [8,14,768]
  float* out_attn = (float*)d_out + 86016;     // [8,14,14]

  // 1) zero accumulators
  zero_kernel<<<(1568 + 255) / 256, 256, 0, stream>>>(adj, 1568);
  zero_kernel<<<(1568 + 255) / 256, 256, 0, stream>>>(attnsum, 1568);

  // 2) cost / fragment / G prep
  {
    int frag5 = 2 * 1 * 32 * 16, tot5 = frag5 + 25 * 25;
    prep_cost_kernel<<<(tot5 + 255) / 256, 256, 0, stream>>>(Ef5, G5, anat, 5, 25, 2, 1);
    int frag15 = 15 * 8 * 32 * 16, tot15 = frag15 + 225 * 225;
    prep_cost_kernel<<<(tot15 + 255) / 256, 256, 0, stream>>>(Ef15, G15, anat, 15, 225, 15, 8);
  }

  // 3) logP / logQ prep
  prep_logpq_kernel<<<(PROBPAD * 32 + 255) / 256, 256, 0, stream>>>(logP5, logQ5, bb, 5, 25, 32);
  prep_logpq_kernel<<<(PROBPAD * 240 + 255) / 256, 256, 0, stream>>>(logP15, logQ15, bb, 15, 225, 240);

  // 4) Sinkhorn (46 WGs x 16 problems, 100 iterations in-kernel, WMMA GEMMs)
  sinkhorn_kernel<<<PROBPAD / 16, 256, 0, stream>>>(Ef5, logP5, logQ5, G5, adj, 25, 2, 1, 0.2f);
  sinkhorn_kernel<<<PROBPAD / 16, 256, 0, stream>>>(Ef15, logP15, logQ15, G15, adj, 225, 15, 8, 0.3f);

  // 5) adjacency softmax
  adj_softmax_kernel<<<1, 128, 0, stream>>>(adj, adjS);

  // 6) embedding GEMM + LN
  gemm_wmma_kernel<<<dim3(7, 16), 32, 0, stream>>>(features, 768, embw, 256, embb, t256, 256, 768);
  ln_kernel<<<112, 256, 0, stream>>>(t256, n1g, n1b, xbuf, 256, 0);

  // 7) GAT layers
  for (int L = 0; L < 3; ++L) {
    const float* wqw = (const float*)d_in[6 + L * 10 + 0];
    const float* wqb = (const float*)d_in[6 + L * 10 + 1];
    const float* wkw = (const float*)d_in[6 + L * 10 + 2];
    const float* wkb = (const float*)d_in[6 + L * 10 + 3];
    const float* wvw = (const float*)d_in[6 + L * 10 + 4];
    const float* wvb = (const float*)d_in[6 + L * 10 + 5];
    const float* wow = (const float*)d_in[6 + L * 10 + 6];
    const float* wob = (const float*)d_in[6 + L * 10 + 7];
    const float* lng = (const float*)d_in[6 + L * 10 + 8];
    const float* lnb = (const float*)d_in[6 + L * 10 + 9];

    gemm_wmma_kernel<<<dim3(7, 16), 32, 0, stream>>>(xbuf, 256, wqw, 256, wqb, qb, 256, 256);
    gemm_wmma_kernel<<<dim3(7, 16), 32, 0, stream>>>(xbuf, 256, wkw, 256, wkb, kb, 256, 256);
    gemm_wmma_kernel<<<dim3(7, 16), 32, 0, stream>>>(xbuf, 256, wvw, 256, wvb, vb, 256, 256);
    gat_attention_kernel<<<8, 256, 0, stream>>>(qb, kb, vb, adjS, attno, attnsum);
    gemm_wmma_kernel<<<dim3(7, 16), 32, 0, stream>>>(attno, 256, wow, 256, wob, t256, 256, 256);
    ln_kernel<<<112, 256, 0, stream>>>(t256, lng, lnb, xbuf, 256, 1);  // LN + ELU
  }

  // 8) projection + final LN -> d_out
  gemm_wmma_kernel<<<dim3(7, 48), 32, 0, stream>>>(xbuf, 256, projw, 768, projb, t768, 768, 256);
  ln_kernel<<<112, 256, 0, stream>>>(t768, n2g, n2b, out_main, 768, 0);

  // 9) final attention -> d_out tail
  copy_kernel<<<(1568 + 255) / 256, 256, 0, stream>>>(attnsum, out_attn, 1568);
}
